// PartialTucker_71648644432003
// MI455X (gfx1250) — matrix operand order repaired and checked by reading, not compiled, not run on hardware
//
#include <hip/hip_runtime.h>

// ---------------------------------------------------------------------------
// PartialTucker on MI455X (gfx1250, wave32, WMMA)
//   ux1 = x1 @ U1   (16384 x 64)      bf16 WMMA, fp32 accumulate
//   ux2 = x2 @ U2   (16384 x 64)      bf16 WMMA, fp32 accumulate
//   Z   = KR(ux1,ux2) @ B1^T          f16 WMMA, fp32 accumulate
// KR is never materialized: A-fragments are rank-1 products built with
// packed v_pk_mul_f16 from fp16 factor tiles in LDS.  Each wave carries
// 4 M-tiles so every B-fragment feeds 4 WMMAs.
// ---------------------------------------------------------------------------

#define OUT_   512
#define D1_    1024
#define R1_    64
#define R2_    64
#define BATCH_ 16384
#define MT_    4          // M-tiles (of 16 rows) per wave

typedef __attribute__((ext_vector_type(16))) __bf16   v16bf;
typedef __attribute__((ext_vector_type(16))) _Float16 v16h;
typedef __attribute__((ext_vector_type(8)))  _Float16 v8h;
typedef __attribute__((ext_vector_type(8)))  float    v8f;

__device__ __forceinline__ v8h splat8(_Float16 x) {
    v8h v;
#pragma unroll
    for (int e = 0; e < 8; ++e) v[e] = x;
    return v;
}

// ---------------------------------------------------------------------------
// Stage 0a: transpose U (1024 x 64 fp32) -> Ut (64 x 1024 bf16).
// ---------------------------------------------------------------------------
__global__ __launch_bounds__(256)
void cvtU_kernel(const float* __restrict__ U1, const float* __restrict__ U2,
                 __bf16* __restrict__ U1t, __bf16* __restrict__ U2t) {
    const float* U  = blockIdx.y ? U2  : U1;
    __bf16*      Ut = blockIdx.y ? U2t : U1t;
    const int idx = blockIdx.x * 256 + threadIdx.x;   // 0 .. 65535
    const int n = idx >> 10;          // 0..63
    const int k = idx & 1023;         // 0..1023
    Ut[(size_t)n * D1_ + k] = (__bf16)U[(size_t)k * R1_ + n];
}

// ---------------------------------------------------------------------------
// Stage 0b: convert B (512 x 4096 fp32 row-major) -> fp16, same layout.
// ---------------------------------------------------------------------------
__global__ __launch_bounds__(256)
void cvtB_kernel(const float* __restrict__ B, _Float16* __restrict__ Bh) {
    const int idx = blockIdx.x * 256 + threadIdx.x;   // 0 .. 2M-1
    Bh[idx] = (_Float16)B[idx];
}

// ---------------------------------------------------------------------------
// Stage 1: projection GEMM  ux = x @ U    (M=16384, N=64, K=1024), bf16.
// 8 waves / block; each wave owns MT_ 16-row M tiles, all 64 N columns, so
// each Ut B-fragment is reused by MT_ WMMAs.  HBM-bound on x (134 MB).
// ---------------------------------------------------------------------------
__global__ __launch_bounds__(256)
void proj_kernel(const float* __restrict__ x1, const float* __restrict__ x2,
                 const __bf16* __restrict__ U1t, const __bf16* __restrict__ U2t,
                 float* __restrict__ ux1, float* __restrict__ ux2) {
    const float*  x  = blockIdx.y ? x2  : x1;
    const __bf16* Ut = blockIdx.y ? U2t : U1t;
    float*        ux = blockIdx.y ? ux2 : ux1;

    const int lane  = threadIdx.x & 31;
    const int wid   = threadIdx.x >> 5;
    const int m0    = (blockIdx.x * 8 + wid) * (16 * MT_);
    const int mrow  = lane & 15;            // A-frag: M index
    const int ncol  = lane & 15;            // B/C-frag: N index
    const int khalf = (lane >> 4) << 3;     // unused (0 / 8); the A-frag also uses kh16
    const int kh16  = (lane >> 4) << 4;     // B-frag K offset (0 / 16)
    const int mhi   = (lane >> 4) << 3;     // C-frag M offset (0 / 8)

    const float* xrow[MT_];
#pragma unroll
    for (int mt = 0; mt < MT_; ++mt)
        xrow[mt] = x + (size_t)(m0 + mt * 16 + mrow) * D1_;

    v8f acc[MT_][4];
#pragma unroll
    for (int mt = 0; mt < MT_; ++mt)
#pragma unroll
        for (int t = 0; t < 4; ++t) acc[mt][t] = v8f{0.f};

    for (int ks = 0; ks < D1_ / 32; ++ks) {
        const int kbase = ks * 32;
        // All 4 B fragments first (clause-paired b128 loads, reused MT_ x).
        v16bf b[4];
#pragma unroll
        for (int t = 0; t < 4; ++t)
            b[t] = *(const v16bf*)(Ut + (size_t)(t * 16 + ncol) * D1_ + kbase + kh16);
#pragma unroll
        for (int mt = 0; mt < MT_; ++mt) {
            const float4* r0 = (const float4*)(xrow[mt] + kbase + kh16);
            const float4* r1 = (const float4*)(xrow[mt] + kbase + kh16 + 8);
            const float4 u0 = r0[0], u1 = r0[1], u2 = r1[0], u3 = r1[1];
            v16bf a;
            a[0]=(__bf16)u0.x;  a[1]=(__bf16)u0.y;  a[2]=(__bf16)u0.z;  a[3]=(__bf16)u0.w;
            a[4]=(__bf16)u1.x;  a[5]=(__bf16)u1.y;  a[6]=(__bf16)u1.z;  a[7]=(__bf16)u1.w;
            a[8]=(__bf16)u2.x;  a[9]=(__bf16)u2.y;  a[10]=(__bf16)u2.z; a[11]=(__bf16)u2.w;
            a[12]=(__bf16)u3.x; a[13]=(__bf16)u3.y; a[14]=(__bf16)u3.z; a[15]=(__bf16)u3.w;
#pragma unroll
            for (int t = 0; t < 4; ++t)
                acc[mt][t] = __builtin_amdgcn_wmma_f32_16x16x32_bf16(
                    false, a, false, b[t], (short)0, acc[mt][t], false, false);
        }
    }

#pragma unroll
    for (int mt = 0; mt < MT_; ++mt)
#pragma unroll
        for (int t = 0; t < 4; ++t)
#pragma unroll
            for (int v = 0; v < 8; ++v)
                ux[(size_t)(m0 + mt * 16 + mhi + v) * R1_ + t * 16 + ncol] = acc[mt][t][v];
}

// ---------------------------------------------------------------------------
// Stage 2: Z = KR(ux1,ux2) @ B1^T   (M=16384, N=512, K=4096), fp16 WMMA.
// One block per 64-row batch tile (MT_ x 16); 8 waves each take a 64-wide N
// slice.  kr[m, i*64+j] = ux1[m,i] * ux2[m,j]; a 32-wide K step has i = ks/2
// fixed and j in [32*(ks&1), +32), so the A-fragment is one splat + two
// v8h packed multiplies per M-tile (v_pk_mul_f16 -> WMMA-ready operand).
// Each B-fragment (2 x b128 from the L2-resident fp16 core) feeds MT_ WMMAs.
// ---------------------------------------------------------------------------
__global__ __launch_bounds__(256)
void tucker_kernel(const float* __restrict__ ux1, const float* __restrict__ ux2,
                   const _Float16* __restrict__ Bh, float* __restrict__ Z) {
    __shared__ _Float16 s1[16 * MT_ * R1_];   // 8 KB
    __shared__ _Float16 s2[16 * MT_ * R2_];   // 8 KB

    const int tid = threadIdx.x;
    const int m0  = blockIdx.x * (16 * MT_);
    for (int t = tid; t < 16 * MT_ * R1_; t += 256) {
        s1[t] = (_Float16)ux1[(size_t)m0 * R1_ + t];
        s2[t] = (_Float16)ux2[(size_t)m0 * R2_ + t];
    }
    __syncthreads();

    const int lane  = tid & 31;
    const int wid   = tid >> 5;
    const int n0    = wid * 64;             // this wave's N slice
    const int mrow  = lane & 15;
    const int ncol  = lane & 15;
    const int khalf = (lane >> 4) << 3;     // 0 / 8
    const int kh16  = (lane >> 4) << 4;     // 0 / 16
    const int mhi   = (lane >> 4) << 3;     // 0 / 8

    v8f acc[MT_][4];
#pragma unroll
    for (int mt = 0; mt < MT_; ++mt)
#pragma unroll
        for (int t = 0; t < 4; ++t) acc[mt][t] = v8f{0.f};

    for (int ks = 0; ks < (R1_ * R2_) / 32; ++ks) {     // 128 steps
        const int i  = ks >> 1;
        const int jb = (ks & 1) * 32;
        // All 4 B fragments first; reused by MT_ M-tiles.
        v16h b[4];
#pragma unroll
        for (int t = 0; t < 4; ++t)
            b[t] = *(const v16h*)(
                Bh + (size_t)(n0 + t * 16 + ncol) * (R1_ * R2_) + ks * 32 + kh16);
#pragma unroll
        for (int mt = 0; mt < MT_; ++mt) {
            const int row = mt * 16 + mrow;
            const v8h av = splat8(s1[row * R1_ + i]);
            const v8h lo = av * *(const v8h*)(&s2[row * R2_ + jb + kh16]);
            const v8h hi = av * *(const v8h*)(&s2[row * R2_ + jb + kh16 + 8]);
            v16h a;
#pragma unroll
            for (int e = 0; e < 8; ++e) { a[e] = lo[e]; a[8 + e] = hi[e]; }
#pragma unroll
            for (int t = 0; t < 4; ++t)
                acc[mt][t] = __builtin_amdgcn_wmma_f32_16x16x32_f16(
                    false, a, false, b[t], (short)0, acc[mt][t], false, false);
        }
    }

#pragma unroll
    for (int mt = 0; mt < MT_; ++mt)
#pragma unroll
        for (int t = 0; t < 4; ++t)
#pragma unroll
            for (int v = 0; v < 8; ++v)
                Z[(size_t)(m0 + mt * 16 + mhi + v) * OUT_ + n0 + t * 16 + ncol]
                    = acc[mt][t][v];
}

// ---------------------------------------------------------------------------
// Launch.  Workspace layout (needs ~12.3 MB):
//   [0,    4MB)  ux1  fp32 16384x64
//   [4MB,  8MB)  ux2  fp32 16384x64
//   [8MB, 12MB)  Bh   fp16 512x4096
//   [12MB, +128KB) U1t bf16 64x1024
//   [.. ,  +128KB) U2t bf16 64x1024
// ---------------------------------------------------------------------------
extern "C" void kernel_launch(void* const* d_in, const int* in_sizes, int n_in,
                              void* d_out, int out_size, void* d_ws, size_t ws_size,
                              hipStream_t stream) {
    const float* x1 = (const float*)d_in[0];
    const float* x2 = (const float*)d_in[1];
    const float* U1 = (const float*)d_in[2];
    const float* U2 = (const float*)d_in[3];
    const float* B  = (const float*)d_in[4];
    float* Z = (float*)d_out;

    char* ws = (char*)d_ws;
    float*    ux1 = (float*)(ws);
    float*    ux2 = (float*)(ws + ((size_t)4 << 20));
    _Float16* Bh  = (_Float16*)(ws + ((size_t)8 << 20));
    __bf16*   U1t = (__bf16*)(ws + ((size_t)12 << 20));
    __bf16*   U2t = (__bf16*)(ws + ((size_t)12 << 20) + ((size_t)128 << 10));

    // Stage 0: precision / layout conversion (tiny, one-off).
    cvtU_kernel<<<dim3(256, 2), 256, 0, stream>>>(U1, U2, U1t, U2t);
    cvtB_kernel<<<dim3((OUT_ * R1_ * R2_) / 256), 256, 0, stream>>>(B, Bh);

    // Stage 1: projections ux1 = x1@U1, ux2 = x2@U2 (bf16 WMMA, fp32 acc).
    proj_kernel<<<dim3(BATCH_ / (16 * MT_ * 8), 2), 256, 0, stream>>>(
        x1, x2, U1t, U2t, ux1, ux2);

    // Stage 2: Z = KR @ B1^T with on-the-fly Khatri-Rao A-fragments (f16 WMMA).
    tucker_kernel<<<dim3(BATCH_ / (16 * MT_)), 256, 0, stream>>>(ux1, ux2, Bh, Z);
}
